// PolygonSegmenter_25890062860394
// MI455X (gfx1250) — compile-verified, matrix operand order checked by reading, not run on hardware
//
#include <hip/hip_runtime.h>
#include <hip/hip_bf16.h>

typedef __attribute__((ext_vector_type(16))) _Float16 v16h;
typedef __attribute__((ext_vector_type(8)))  float    v8f;

#define NN      100000
#define NE      800000
#define NL      262144
#define F_IN    16
#define F_H     256
#define F_OUT   128
#define BN_EPS  1e-5f

// fragment-layout helper (ISA 7.12.2, 16-bit A/B 16x32): K offset for
// (vgpr v, lane-half hsel) within a 32-wide k-block
__device__ __forceinline__ int frag_kbase(int v, int hsel) {
    return ((v < 4) ? 0 : 16) + (hsel ? 8 : 0) + 2 * (v & 3);
}

// ---------------------------------------------------------------- utilities
__global__ void zero_f32(float* __restrict__ p, int n) {
    int i = blockIdx.x * blockDim.x + threadIdx.x;
    if (i < n) p[i] = 0.0f;
}

// deg[dst] += w
__global__ void deg_accum(const int* __restrict__ dst, const float* __restrict__ w,
                          float* __restrict__ deg, int n) {
    int i = blockIdx.x * blockDim.x + threadIdx.x;
    if (i < n) atomicAdd(&deg[dst[i]], w[i]);
}

__global__ void deg_rsqrt(float* __restrict__ p, int n) {
    int i = blockIdx.x * blockDim.x + threadIdx.x;
    if (i < n) p[i] = rsqrtf(p[i] + 1.0f);
}

__global__ void edge_norm_k(const int* __restrict__ src, const int* __restrict__ dst,
                            const float* __restrict__ w, const float* __restrict__ dinv,
                            float* __restrict__ en, int n) {
    int i = blockIdx.x * blockDim.x + threadIdx.x;
    if (i < n) en[i] = dinv[src[i]] * w[i] * dinv[dst[i]];
}

// -------------------------------------------- pack W (f32, KxN) into B-fragment
// order: Bp[(kb * (Nout/16) + nt)*512 + lane*16 + h]; zero-pad k >= Ksrc.
__global__ void pack_b(const float* __restrict__ Wsrc, _Float16* __restrict__ Bp,
                       int Ksrc, int Kpad, int Nout) {
    int i = blockIdx.x * blockDim.x + threadIdx.x;
    int total = Kpad * Nout;
    if (i >= total) return;
    int h    = i & 15;
    int lane = (i >> 4) & 31;
    int tile = i >> 9;                 // kb * ntiles + nt
    int ntiles = Nout >> 4;
    int kb = tile / ntiles;
    int nt = tile - kb * ntiles;
    int v = h >> 1, odd = h & 1;
    int hsel = lane >> 4;
    int k = kb * 32 + frag_kbase(v, hsel) + odd;
    int n = nt * 16 + (lane & 15);
    Bp[i] = (k < Ksrc) ? (_Float16)Wsrc[(size_t)k * Nout + n] : (_Float16)0.0f;
}

// ------------------------------------------------- WMMA GEMM (f16 in, f32 acc)
// Y[M x Nout] = A[M x Ksrc] * W (+bias)(+relu); KPAD = Ksrc rounded to 32.
// Block = 128 threads (4 waves); wave w -> 16x16 tile at col blockIdx.y*64+w*16.
// A staged in LDS directly in fragment order -> inner loop is
//   2x ds_load_b128 + 2x global_load_b128 + v_wmma, no guards.
template <int KPAD>
__global__ void wmma_gemm(const float* __restrict__ X,
                          const _Float16* __restrict__ Bp,
                          const float* __restrict__ bias,
                          float* __restrict__ Y,
                          int Mrows, int Ksrc, int Nout,
                          const int* __restrict__ ge0,
                          const int* __restrict__ ge1,
                          int do_relu) {
    constexpr int KBLK = KPAD / 32;
    __shared__ _Float16 As[KBLK * 512];          // fragment-ordered A tile

    const int tid  = threadIdx.x;
    const int wave = tid >> 5;
    const int lane = tid & 31;
    const int rowbase = blockIdx.x * 16;
    const int nt      = blockIdx.y * 4 + wave;   // 16-col tile index
    const int ntiles  = Nout >> 4;

    // stage 16 x KPAD A tile into LDS in fragment order
    for (int j = tid; j < KBLK * 512; j += 128) {
        int h    = j & 15;
        int l    = (j >> 4) & 31;
        int kblk = j >> 9;
        int v = h >> 1, odd = h & 1;
        int hsel = l >> 4;
        int m    = l & 15;
        int k    = kblk * 32 + frag_kbase(v, hsel) + odd;
        int row  = rowbase + m;
        float val = 0.0f;
        if (row < Mrows && k < Ksrc) {
            if (ge0) {   // decoder: A row = concat(z[e0], z[e1]), each F_OUT wide
                val = (k < F_OUT) ? X[(size_t)ge0[row] * F_OUT + k]
                                  : X[(size_t)ge1[row] * F_OUT + (k - F_OUT)];
            } else {
                val = X[(size_t)row * Ksrc + k];
            }
        }
        As[j] = (_Float16)val;
    }
    __syncthreads();

    v8f acc = {};
    const _Float16* __restrict__ bptr = Bp + ((size_t)nt * 512 + lane * 16);
    const _Float16* __restrict__ aptr = As + lane * 16;
    const size_t bstride = (size_t)ntiles * 512;  // next k-block
#pragma unroll
    for (int kb = 0; kb < KBLK; ++kb) {
        v16h a = *(const v16h*)(aptr + kb * 512);
        v16h b = *(const v16h*)(bptr + kb * bstride);
        acc = __builtin_amdgcn_wmma_f32_16x16x32_f16(
            false, a, false, b, (short)0, acc, false, false);
    }

    // D layout: VGPR v -> M = v (+8 for lanes 16-31), N = lane&15
    const int hsel = lane >> 4;
    const int n    = lane & 15;
    const int col  = nt * 16 + n;
    float bval = bias ? bias[col] : 0.0f;
#pragma unroll
    for (int v = 0; v < 8; ++v) {
        int row = rowbase + v + (hsel ? 8 : 0);
        if (row < Mrows) {
            float val = acc[v] + bval;
            if (do_relu) val = fmaxf(val, 0.0f);
            Y[(size_t)row * Nout + col] = val;
        }
    }
}

// out = hlin * dinv[row]^2 + b[col]    (self-loop term + bias, pre-scatter)
__global__ void out_init(const float* __restrict__ hlin, const float* __restrict__ dinv,
                         const float* __restrict__ b, float* __restrict__ out,
                         int nrows, int F) {
    size_t i = (size_t)blockIdx.x * blockDim.x + threadIdx.x;
    size_t total = (size_t)nrows * F;
    if (i < total) {
        int r = (int)(i / F);
        int f = (int)(i - (size_t)r * F);
        float dv = dinv[r];
        out[i] = hlin[i] * dv * dv + b[f];
    }
}

// wave-per-edge scatter: out[dst] += h[src] * edge_norm (float4 gather, f32 atomics)
__global__ void scatter_edges(const float4* __restrict__ hlin,
                              const int* __restrict__ src, const int* __restrict__ dst,
                              const float* __restrict__ en,
                              float* __restrict__ out, int nedges, int F) {
    int gw   = (blockIdx.x * blockDim.x + threadIdx.x) >> 5;
    int lane = threadIdx.x & 31;
    int nw   = (gridDim.x * blockDim.x) >> 5;
    int Fq   = F >> 2;
    for (int e = gw; e < nedges; e += nw) {
        int s = src[e], d = dst[e];
        float w = en[e];
        for (int c = lane; c < Fq; c += 32) {
            float4 v = hlin[(size_t)s * Fq + c];
            float* o = out + (size_t)d * F + (size_t)c * 4;
            atomicAdd(o + 0, v.x * w);
            atomicAdd(o + 1, v.y * w);
            atomicAdd(o + 2, v.z * w);
            atomicAdd(o + 3, v.w * w);
        }
    }
}

// per-feature sum / sumsq (F==256): lane f strides rows; coalesced reads
#define BN_ROWS 512
__global__ void bn_stats(const float* __restrict__ h, float* __restrict__ sums, int nrows) {
    int f  = threadIdx.x;                        // 0..255
    int r0 = blockIdx.x * BN_ROWS;
    float s = 0.0f, ss = 0.0f;
    int rend = min(r0 + BN_ROWS, nrows);
    for (int r = r0; r < rend; ++r) {
        float v = h[(size_t)r * F_H + f];
        s += v; ss += v * v;
    }
    atomicAdd(&sums[f], s);
    atomicAdd(&sums[F_H + f], ss);
}

__global__ void bn_apply_relu(const float* __restrict__ x, const float* __restrict__ sums,
                              const float* __restrict__ g, const float* __restrict__ be,
                              float* __restrict__ y, int nrows) {
    size_t i = (size_t)blockIdx.x * blockDim.x + threadIdx.x;
    size_t total = (size_t)nrows * F_H;
    if (i < total) {
        int f = (int)(i & (F_H - 1));
        float m  = sums[f] * (1.0f / NN);
        float v  = sums[F_H + f] * (1.0f / NN) - m * m;
        float o  = (x[i] - m) * rsqrtf(v + BN_EPS) * g[f] + be[f];
        y[i] = fmaxf(o, 0.0f);
    }
}

// final decoder layer: p = sigmoid(q . D3W + D3b); wave per row, shuffle reduce
__global__ void dec_final(const float* __restrict__ q, const float* __restrict__ w,
                          const float* __restrict__ b, float* __restrict__ out, int M) {
    int wave = (blockIdx.x * blockDim.x + threadIdx.x) >> 5;
    int lane = threadIdx.x & 31;
    if (wave >= M) return;
    const float* row = q + (size_t)wave * F_OUT;
    float s = 0.0f;
    for (int k = lane; k < F_OUT; k += 32) s += row[k] * w[k];
    for (int off = 16; off > 0; off >>= 1) s += __shfl_xor(s, off, 32);
    if (lane == 0) out[wave] = 1.0f / (1.0f + __expf(-(s + b[0])));
}

// ---------------------------------------------------------------- launcher
extern "C" void kernel_launch(void* const* d_in, const int* in_sizes, int n_in,
                              void* d_out, int out_size, void* d_ws, size_t ws_size,
                              hipStream_t stream) {
    const float* x   = (const float*)d_in[0];
    const int*   ei  = (const int*)d_in[1];
    const float* ew  = (const float*)d_in[2];
    const int*   eli = (const int*)d_in[3];
    const float* W1 = (const float*)d_in[4],  *b1 = (const float*)d_in[5];
    const float* g1 = (const float*)d_in[6],  *be1 = (const float*)d_in[7];
    const float* W2 = (const float*)d_in[8],  *b2 = (const float*)d_in[9];
    const float* g2 = (const float*)d_in[10], *be2 = (const float*)d_in[11];
    const float* W3 = (const float*)d_in[12], *b3 = (const float*)d_in[13];
    const float* g3 = (const float*)d_in[14], *be3 = (const float*)d_in[15];
    const float* W4 = (const float*)d_in[16], *b4 = (const float*)d_in[17];
    const float* D1W = (const float*)d_in[18], *D1b = (const float*)d_in[19];
    const float* D2W = (const float*)d_in[20], *D2b = (const float*)d_in[21];
    const float* D3W = (const float*)d_in[22], *D3b = (const float*)d_in[23];
    float* p_out = (float*)d_out;

    const int* e_src = ei;            // edge_index[0]
    const int* e_dst = ei + NE;       // edge_index[1]
    const int* l_e0  = eli;           // edge_label_index[0]
    const int* l_e1  = eli + NL;

    // ---- workspace carve-up (float units, 256B aligned chunks)
    float* W = (float*)d_ws;
    size_t o = 0;
    auto carve = [&](size_t nfloats) -> float* {
        float* p = W + o;
        o += (nfloats + 63) & ~(size_t)63;
        return p;
    };
    float*     dinv  = carve(NN);
    float*     enorm = carve(NE);
    float*     sums  = carve(2 * F_H);
    _Float16*  W1p   = (_Float16*)carve(32 * F_H);          // K padded 16->32
    _Float16*  W2p   = (_Float16*)carve(F_H * F_H);
    _Float16*  W3p   = (_Float16*)carve(F_H * F_H);
    _Float16*  W4p   = (_Float16*)carve(F_H * F_OUT);
    _Float16*  D1Wp  = (_Float16*)carve(2 * F_OUT * F_OUT);
    _Float16*  D2Wp  = (_Float16*)carve(F_OUT * F_OUT);
    float*     buf0  = carve((size_t)NN * F_H);   // bn output / layer input
    float*     buf1  = carve((size_t)NN * F_H);   // hlin = h @ W
    float*     buf2  = carve((size_t)NN * F_H);   // aggregated layer output
    float*     q1    = carve((size_t)NL * F_OUT);
    float*     q2    = carve((size_t)NL * F_OUT);

    auto cdiv = [](long a, long b) { return (int)((a + b - 1) / b); };

    // ---- normalization precompute
    zero_f32<<<cdiv(NN, 256), 256, 0, stream>>>(dinv, NN);
    deg_accum<<<cdiv(NE, 256), 256, 0, stream>>>(e_dst, ew, dinv, NE);
    deg_rsqrt<<<cdiv(NN, 256), 256, 0, stream>>>(dinv, NN);
    edge_norm_k<<<cdiv(NE, 256), 256, 0, stream>>>(e_src, e_dst, ew, dinv, enorm, NE);

    // ---- weights -> f16 fragment layout (once per launch)
    pack_b<<<cdiv(32 * F_H, 256), 256, 0, stream>>>(W1, W1p, F_IN, 32, F_H);
    pack_b<<<cdiv(F_H * F_H, 256), 256, 0, stream>>>(W2, W2p, F_H, F_H, F_H);
    pack_b<<<cdiv(F_H * F_H, 256), 256, 0, stream>>>(W3, W3p, F_H, F_H, F_H);
    pack_b<<<cdiv(F_H * F_OUT, 256), 256, 0, stream>>>(W4, W4p, F_H, F_H, F_OUT);
    pack_b<<<cdiv(2 * F_OUT * F_OUT, 256), 256, 0, stream>>>(D1W, D1Wp, 2 * F_OUT,
                                                             2 * F_OUT, F_OUT);
    pack_b<<<cdiv(F_OUT * F_OUT, 256), 256, 0, stream>>>(D2W, D2Wp, F_OUT, F_OUT, F_OUT);

    // ---- GCN layers
    auto layer_tail = [&](const float* bias, const float* g, const float* be,
                          int Fout, bool do_bn) {
        size_t tot = (size_t)NN * Fout;
        out_init<<<cdiv((long)tot, 256), 256, 0, stream>>>(buf1, dinv, bias, buf2, NN, Fout);
        scatter_edges<<<2048, 256, 0, stream>>>((const float4*)buf1, e_src, e_dst,
                                                enorm, buf2, NE, Fout);
        if (do_bn) {
            zero_f32<<<2, 256, 0, stream>>>(sums, 2 * F_H);
            bn_stats<<<cdiv(NN, BN_ROWS), F_H, 0, stream>>>(buf2, sums, NN);
            bn_apply_relu<<<cdiv((long)tot, 256), 256, 0, stream>>>(buf2, sums, g, be,
                                                                    buf0, NN);
        }
    };

    dim3 gh(cdiv(NN, 16), F_H / 64);
    dim3 go(cdiv(NN, 16), F_OUT / 64);

    wmma_gemm<32><<<gh, 128, 0, stream>>>(x, W1p, nullptr, buf1, NN, F_IN, F_H,
                                          nullptr, nullptr, 0);
    layer_tail(b1, g1, be1, F_H, true);

    wmma_gemm<256><<<gh, 128, 0, stream>>>(buf0, W2p, nullptr, buf1, NN, F_H, F_H,
                                           nullptr, nullptr, 0);
    layer_tail(b2, g2, be2, F_H, true);

    wmma_gemm<256><<<gh, 128, 0, stream>>>(buf0, W3p, nullptr, buf1, NN, F_H, F_H,
                                           nullptr, nullptr, 0);
    layer_tail(b3, g3, be3, F_H, true);

    wmma_gemm<256><<<go, 128, 0, stream>>>(buf0, W4p, nullptr, buf1, NN, F_H, F_OUT,
                                           nullptr, nullptr, 0);
    layer_tail(b4, nullptr, nullptr, F_OUT, false);   // z in buf2

    // ---- decoder: concat-gather fused into WMMA A staging
    dim3 gd(cdiv(NL, 16), F_OUT / 64);
    wmma_gemm<256><<<gd, 128, 0, stream>>>(buf2, D1Wp, D1b, q1, NL, 2 * F_OUT, F_OUT,
                                           l_e0, l_e1, 1);
    wmma_gemm<128><<<gd, 128, 0, stream>>>(q1, D2Wp, D2b, q2, NL, F_OUT, F_OUT,
                                           nullptr, nullptr, 1);
    dec_final<<<cdiv(NL, 8), 256, 0, stream>>>(q2, D3W, D3b, p_out, NL);
}
